// HighwayLayer_47571057770809
// MI455X (gfx1250) — matrix operand order repaired and checked by reading, not compiled, not run on hardware
//
#include <hip/hip_runtime.h>

// ---------------------------------------------------------------------------
// Recurrent highway net, MI455X (gfx1250):
//   pack_w:  f32 weights -> bf16 *fragment-major* packs (each lane's 16
//            values contiguous -> B-fragment = one 32B load, no scalar gathers)
//   xproj:   xproj[t][btile] = x[:,t,:] @ W_x + b_in, fully parallel WMMA
//            GEMM; output stored in WMMA C-fragment order so the scan reads
//            one b128 per lane per step.
//   scan:    persistent, 8 WGs (batch rows independent -> no grid sync).
//            Weights pinned in VGPRs, activations ping-pong in LDS (bf16),
//            residual y carried in f32 registers.
// ---------------------------------------------------------------------------

typedef __bf16 bf16_t;
typedef __attribute__((ext_vector_type(16))) __bf16 v16bf;
typedef __attribute__((ext_vector_type(8)))  __bf16 v8bf;
typedef __attribute__((ext_vector_type(4)))  __bf16 v4bf;
typedef __attribute__((ext_vector_type(8)))  float  v8f;
typedef __attribute__((ext_vector_type(4)))  float  v4f;

#define B_    128
#define T_    512
#define U_    256
#define LRA   0.2f
#define LDA_  264   // padded LDS row (264*2B = 528B; 16B-aligned rows)

// ---- workspace layout (bytes); total ~34.3 MB ------------------------------
#define WX_OFF     0u
#define WY_OFF     131072u
#define WH_OFF     262144u        // 2 packed matrices
#define WOUT_OFF   524288u
#define XPROJ_OFF  655360u        // 512*8*16*32*8 bf16 = 33,554,432 B

static __device__ __forceinline__ bf16_t f2bf(float f) {
  unsigned u = __builtin_bit_cast(unsigned, f);
  unsigned r = u + 0x7FFFu + ((u >> 16) & 1u);       // round-to-nearest-even
  return __builtin_bit_cast(bf16_t, (unsigned short)(r >> 16));
}
static __device__ __forceinline__ float bf2f(bf16_t b) {
  unsigned short s = __builtin_bit_cast(unsigned short, b);
  return __builtin_bit_cast(float, ((unsigned)s) << 16);
}

static __device__ __forceinline__ v8f wmma_bf16(v16bf a, v16bf b, v8f c) {
  return __builtin_amdgcn_wmma_f32_16x16x32_bf16(false, a, false, b,
                                                 (short)0, c, false, false);
}

// A fragment: 16x32 bf16 from LDS (ISA 7.12.2 16-bit A layout).
// Two aligned 16B ds loads, merged with a shuffle.
static __device__ __forceinline__ v16bf load_a_frag(const bf16_t* A, int lda,
                                                    int kbase, int lane) {
  const int m = lane & 15;
  const int g = lane >> 4;
  const bf16_t* row = A + m * lda + kbase + g * 8;
  v8bf lo = *(const v8bf*)(row);        // K rel 0..7
  v8bf hi = *(const v8bf*)(row + 16);   // K rel 16..23
  return __builtin_shufflevector(lo, hi, 0, 1, 2, 3, 4, 5, 6, 7,
                                 8, 9, 10, 11, 12, 13, 14, 15);
}

// B fragment from fragment-major pack: one contiguous 32B load per lane.
static __device__ __forceinline__ v16bf load_b_frag_packed(const bf16_t* Wp,
                                                           int nt, int ks,
                                                           int lane) {
  const bf16_t* p = Wp + ((size_t)(((nt * 8 + ks) * 32) + lane) << 4);
  return *(const v16bf*)(p);
}

// xproj fragment address: [t][btile][ntile][lane][8] bf16
static __device__ __forceinline__ size_t xp_off(int t, int bt, int nt,
                                                int lane) {
  return ((((size_t)t * (B_ / 16) + bt) * 16 + nt) * 32 + lane) * 8;
}

// ---------------------------------------------------------------------------
// Pack one 256x256 f32 matrix into fragment-major bf16:
// dst[((nt*8+ks)*32+lane)*16 + j] = src[ks*32 + 8*(lane>>4) + j + (j&8)][nt*16+(lane&15)]
__global__ void pack_w_kernel(const float* __restrict__ src,
                              bf16_t* __restrict__ dst) {
  int id = blockIdx.x * blockDim.x + threadIdx.x;  // 0..4095
  int lane = id & 31;
  int ks = (id >> 5) & 7;
  int nt = id >> 8;
  int n = nt * 16 + (lane & 15);
  int k0 = ks * 32 + 8 * (lane >> 4);
  bf16_t* d = dst + ((size_t)id << 4);
#pragma unroll
  for (int j = 0; j < 16; ++j) d[j] = f2bf(src[(size_t)(k0 + j + (j & 8)) * U_ + n]);
}

// ---------------------------------------------------------------------------
// xproj GEMM. Block = (t, btile): 16 batch rows at one timestep so the
// output tile matches the scan's consumer tile exactly.
__global__ __launch_bounds__(512, 1) void xproj_kernel(
    const float* __restrict__ x, const bf16_t* __restrict__ Wxp,
    const float* __restrict__ b_in, bf16_t* __restrict__ xproj) {
  __shared__ __align__(16) bf16_t As[16][LDA_];
  const int t    = blockIdx.x >> 3;   // 0..511
  const int bt   = blockIdx.x & 7;    // batch tile
  const int b0   = bt * 16;
  const int tid  = threadIdx.x;
  const int lane = tid & 31;
  const int wv   = tid >> 5;          // ntile 0..15

  // stage A = bf16(x[b0..b0+15, t, :]) ; vectorized f32x4 -> bf16x4
  for (int idx = tid; idx < 16 * (U_ / 4); idx += 512) {
    int m = idx >> 6, k4 = (idx & 63) * 4;
    v4f xv = *(const v4f*)(x + ((size_t)(b0 + m) * T_ + t) * U_ + k4);
    v4bf o;
#pragma unroll
    for (int j = 0; j < 4; ++j) o[j] = f2bf(xv[j]);
    *(v4bf*)(&As[m][k4]) = o;
  }
  __syncthreads();

  v8f acc = {};
#pragma unroll
  for (int s = 0; s < 8; ++s)
    acc = wmma_bf16(load_a_frag(&As[0][0], LDA_, s * 32, lane),
                    load_b_frag_packed(Wxp, wv, s, lane), acc);

  const float bi = b_in[wv * 16 + (lane & 15)];
  v8bf o;
#pragma unroll
  for (int i = 0; i < 8; ++i) o[i] = f2bf(acc[i] + bi);
  *(v8bf*)(xproj + xp_off(t, bt, wv, lane)) = o;   // one 16B store per lane
}

// ---------------------------------------------------------------------------
// Persistent scan. grid = 8 WGs x 1024 threads (32 waves).
// half 0 (waves 0-15):  layer1 (W_y), layer2 (W_h[0])
// half 1 (waves 16-31): layer3 (W_h[1]), layer4 (W_out)
__global__ __launch_bounds__(1024, 1) void rhw_scan_kernel(
    const bf16_t* __restrict__ Wyp, const bf16_t* __restrict__ Whp,
    const bf16_t* __restrict__ Woutp,
    const float* __restrict__ b_h, const float* __restrict__ b_out,
    const float* __restrict__ h0, const bf16_t* __restrict__ xproj,
    float* __restrict__ out) {
  __shared__ __align__(16) bf16_t abuf[4][16][LDA_];  // buf3 = y, buf0..2 = h1..h3

  const int bt   = blockIdx.x;        // batch tile 0..7
  const int m0   = bt * 16;
  const int tid  = threadIdx.x;
  const int wave = tid >> 5;
  const int lane = tid & 31;
  const int half = wave >> 4;
  const int nt   = wave & 15;
  const int n0   = nt * 16;
  const int ncol = lane & 15;
  const int g    = lane >> 4;

  // --- pin weight fragments in VGPRs (one-time, 2x b128 per fragment) ---
  const bf16_t* W0 = (half == 0) ? Wyp : (Whp + U_ * U_);
  const bf16_t* W1 = (half == 0) ? Whp : Woutp;
  v16bf wf0[8], wf1[8];
#pragma unroll
  for (int s = 0; s < 8; ++s) {
    wf0[s] = load_b_frag_packed(W0, nt, s, lane);
    wf1[s] = load_b_frag_packed(W1, nt, s, lane);
  }

  // --- y_0 = h0 broadcast ---
  for (int idx = tid; idx < 16 * U_; idx += 1024) {
    int m = idx >> 8, u = idx & (U_ - 1);
    abuf[3][m][u] = f2bf(h0[u]);
  }
  v8f yreg;
#pragma unroll
  for (int i = 0; i < 8; ++i) yreg[i] = h0[n0 + ncol];
  __syncthreads();

  for (int t = 0; t < T_; ++t) {
    // ---- phase 1: h1 = lrelu(y @ W_y + xproj[t])  (buf3 -> buf0)
    if (half == 0) {
      v8f acc = {};
#pragma unroll
      for (int s = 0; s < 8; ++s)
        acc = wmma_bf16(load_a_frag(&abuf[3][0][0], LDA_, s * 32, lane),
                        wf0[s], acc);
      v8bf xv = *(const v8bf*)(xproj + xp_off(t, bt, nt, lane));
#pragma unroll
      for (int i = 0; i < 8; ++i) {
        float v = acc[i] + bf2f(xv[i]);
        v = v > 0.f ? v : LRA * v;
        abuf[0][i + 8 * g][n0 + ncol] = f2bf(v);
      }
      if (t + 1 < T_)  // pull next step's fragment line toward the WGP
        __builtin_prefetch(xproj + xp_off(t + 1, bt, nt, lane), 0, 0);
    }
    __syncthreads();
    // ---- phase 2: h2 = lrelu(h1 @ W_h0 + b_h0)  (buf0 -> buf1)
    if (half == 0) {
      v8f acc = {};
#pragma unroll
      for (int s = 0; s < 8; ++s)
        acc = wmma_bf16(load_a_frag(&abuf[0][0][0], LDA_, s * 32, lane),
                        wf1[s], acc);
      const float bh0 = b_h[n0 + ncol];
#pragma unroll
      for (int i = 0; i < 8; ++i) {
        float v = acc[i] + bh0;
        v = v > 0.f ? v : LRA * v;
        abuf[1][i + 8 * g][n0 + ncol] = f2bf(v);
      }
    }
    __syncthreads();
    // ---- phase 3: h3 = lrelu(h2 @ W_h1 + b_h1)  (buf1 -> buf2)
    if (half == 1) {
      v8f acc = {};
#pragma unroll
      for (int s = 0; s < 8; ++s)
        acc = wmma_bf16(load_a_frag(&abuf[1][0][0], LDA_, s * 32, lane),
                        wf0[s], acc);
      const float bh1 = b_h[U_ + n0 + ncol];
#pragma unroll
      for (int i = 0; i < 8; ++i) {
        float v = acc[i] + bh1;
        v = v > 0.f ? v : LRA * v;
        abuf[2][i + 8 * g][n0 + ncol] = f2bf(v);
      }
    }
    __syncthreads();
    // ---- phase 4: y_t = h3 @ W_out + b_out + y_{t-1}  (buf2 -> buf3 + out)
    if (half == 1) {
      v8f acc = {};
#pragma unroll
      for (int s = 0; s < 8; ++s)
        acc = wmma_bf16(load_a_frag(&abuf[2][0][0], LDA_, s * 32, lane),
                        wf1[s], acc);
      const float bo = b_out[n0 + ncol];
#pragma unroll
      for (int i = 0; i < 8; ++i) {
        int m = i + 8 * g;
        float v = acc[i] + bo + yreg[i];
        yreg[i] = v;
        out[((size_t)(m0 + m) * T_ + t) * U_ + n0 + ncol] = v;
        abuf[3][m][n0 + ncol] = f2bf(v);
      }
    }
    __syncthreads();
  }
}

// ---------------------------------------------------------------------------
extern "C" void kernel_launch(void* const* d_in, const int* in_sizes, int n_in,
                              void* d_out, int out_size, void* d_ws,
                              size_t ws_size, hipStream_t stream) {
  const float* x     = (const float*)d_in[0];  // [128,512,256]
  const float* W_y   = (const float*)d_in[1];  // [256,256]
  const float* W_x   = (const float*)d_in[2];  // [256,256]
  const float* b_in  = (const float*)d_in[3];  // [256]
  const float* W_h   = (const float*)d_in[4];  // [2,256,256]
  const float* b_h   = (const float*)d_in[5];  // [2,256]
  const float* W_out = (const float*)d_in[6];  // [256,256]
  const float* b_out = (const float*)d_in[7];  // [256]
  const float* h0    = (const float*)d_in[8];  // [1,256]
  float* out = (float*)d_out;

  char* ws = (char*)d_ws;
  bf16_t* wxp   = (bf16_t*)(ws + WX_OFF);
  bf16_t* wyp   = (bf16_t*)(ws + WY_OFF);
  bf16_t* whp   = (bf16_t*)(ws + WH_OFF);
  bf16_t* woutp = (bf16_t*)(ws + WOUT_OFF);
  bf16_t* xp    = (bf16_t*)(ws + XPROJ_OFF);

  pack_w_kernel<<<16, 256, 0, stream>>>(W_x, wxp);
  pack_w_kernel<<<16, 256, 0, stream>>>(W_y, wyp);
  pack_w_kernel<<<16, 256, 0, stream>>>(W_h, whp);
  pack_w_kernel<<<16, 256, 0, stream>>>(W_h + U_ * U_, whp + U_ * U_);
  pack_w_kernel<<<16, 256, 0, stream>>>(W_out, woutp);

  xproj_kernel<<<T_ * (B_ / 16), 512, 0, stream>>>(x, wxp, b_in, xp);

  rhw_scan_kernel<<<B_ / 16, 1024, 0, stream>>>(wyp, whp, woutp, b_h, b_out,
                                                h0, xp, out);
}
